// Attention_66108136620268
// MI455X (gfx1250) — matrix operand order repaired and checked by reading, not compiled, not run on hardware
//
#include <hip/hip_runtime.h>

typedef _Float16 h16;
typedef __attribute__((ext_vector_type(16))) _Float16 v16h;
typedef __attribute__((ext_vector_type(8)))  _Float16 v8h;
typedef __attribute__((ext_vector_type(8)))  float    v8f;

#define Bt   32
#define Nt   512
#define Ct   768
#define Ht   12
#define HDt  64
#define SCALEF 0.125f     // 64^-0.5
#define BOUND 358         // ceil(0.7 * 511)

union AFrag { v16h v; h16 h[16]; };

// A-fragment (16x32 f16), CDNA5 layout:
// lanes 0-15: M=lane, halves0-7 = K 0..7,  halves8-15 = K 16..23
// lanes16-31: M=lane-16, halves0-7 = K 8..15, halves8-15 = K 24..31
__device__ __forceinline__ v16h load_a16(const h16* base, int lda, int koff, int lane) {
  int row = lane & 15;
  int kh  = (lane >> 4) << 3;
  const h16* p = base + row * lda + koff + kh;
  AFrag a;
  *reinterpret_cast<v8h*>(&a.h[0]) = *reinterpret_cast<const v8h*>(p);
  *reinterpret_cast<v8h*>(&a.h[8]) = *reinterpret_cast<const v8h*>(p + 16);
  return a.v;
}

// Same A layout but source is f32 (softmax probabilities in LDS), convert on the fly.
__device__ __forceinline__ v16h load_a32(const float* base, int lda, int koff, int lane) {
  int row = lane & 15;
  int kh  = (lane >> 4) << 3;
  const float* p0 = base + row * lda + koff + kh;
  const float* p1 = p0 + 16;
  AFrag a;
#pragma unroll
  for (int i = 0; i < 8; i++) { a.h[i] = (h16)p0[i]; a.h[8 + i] = (h16)p1[i]; }
  return a.v;
}

// B-fragment (32x16 f16): N = lane&15; halves0-7 = K 8h..8h+7, halves8-15 = K 16+8h..23+8h (h = lane>>4).
__device__ __forceinline__ v16h load_b16(const h16* base, int ldb, int lane) {
  int col = lane & 15;
  int kh  = (lane >> 4) << 3;
  AFrag b;
#pragma unroll
  for (int i = 0; i < 8; i++) {
    b.h[i]     = base[(kh + i) * ldb + col];
    b.h[8 + i] = base[(16 + kh + i) * ldb + col];
  }
  return b.v;
}

__device__ __forceinline__ v8f wmma16(v16h a, v16h b, v8f c) {
  return __builtin_amdgcn_wmma_f32_16x16x32_f16(false, a, false, b, (short)0, c, false, false);
}

// ---------------- Tensor Data Mover: contiguous 1-D tile -> LDS ----------------
#if __has_builtin(__builtin_amdgcn_tensor_load_to_lds)
#define HAVE_TDM 1
typedef unsigned int v4u __attribute__((ext_vector_type(4)));
typedef int          v8i __attribute__((ext_vector_type(8)));
typedef int          v4i __attribute__((ext_vector_type(4)));

// Issue a TDM load of `bytes` (multiple of 8, <= 8192*8) contiguous bytes from
// global `gptr` into LDS byte offset `lds_off`. One instruction per wave
// (EXEC ignored); tracked by TENSORcnt.
__device__ __forceinline__ void tdm_load_1d(unsigned lds_off, const void* gptr, unsigned bytes) {
  unsigned long long ga = (unsigned long long)gptr;
  unsigned elems = bytes >> 3;                       // 8-byte elements
  v4u g0;
  g0[0] = 1u;                                        // count=1, user descriptor
  g0[1] = lds_off;                                   // lds_addr (bytes)
  g0[2] = (unsigned)ga;                              // global_addr[31:0]
  g0[3] = (unsigned)((ga >> 32) & 0x01ffffffu) | (2u << 30);  // addr[56:32] | type=2
  v8i g1;
  g1[0] = 3 << 16;                                   // data_size = 8B
  g1[1] = (int)(elems << 16);                        // tensor_dim0[15:0] @ bits[63:48]
  g1[2] = (int)(((elems >> 16) & 0xffffu) | (1u << 16)); // dim0 hi | tensor_dim1 = 1
  g1[3] = (int)(elems << 16);                        // tile_dim0 @ bits[127:112]
  g1[4] = 1;                                         // tile_dim1 = 1, tile_dim2 = 0
  g1[5] = (int)elems;                                // tensor_dim0_stride low 32
  g1[6] = 0;                                         // stride hi, dim1_stride lo
  g1[7] = 0;                                         // dim1_stride hi
  v4i z4 = {};
  v8i z8 = {};
  // 6-arg form (clang-23 / therock-10.0 headers): (g0, g1, g2, g3, extra, cpol)
  __builtin_amdgcn_tensor_load_to_lds(g0, g1, z4, z4, z8, 0);
}
#else
#define HAVE_TDM 0
#endif

// ---------------- elementwise f32 -> f16 ----------------
__global__ void cvt_f16_kernel(const float* __restrict__ src, h16* __restrict__ dst, int n) {
  int i = blockIdx.x * blockDim.x + threadIdx.x;
  if (i < n) dst[i] = (h16)src[i];
}

// ---------------- tiled WMMA GEMM (128x128x32), 8 waves (2x4) ----------------
// mode 0: A=xh [M,768], B=wqkv [768,2304]; scatter Q[B,H,N,HD], Kt[B,H,HD,N], V[B,H,N,HD] (f16)
// mode 1: A=Oh [M,768], B=wproj [768,768]; Cout = A@B + bias (f32)
__global__ __launch_bounds__(256) void gemm_kernel(
    const h16* __restrict__ A, const h16* __restrict__ Bm,
    int M, int Nc, int K, int mode,
    h16* __restrict__ Qh, h16* __restrict__ Ktp, h16* __restrict__ Vh,
    const float* __restrict__ bias, float* __restrict__ Cout) {
  __shared__ __align__(32) h16 As[128][32];
  __shared__ __align__(32) h16 Bs[32][128];
  int tid = threadIdx.x, lane = tid & 31, w = tid >> 5;
  int wm = w >> 2, wn = w & 3;                 // 2 x 4 wave grid
  int bm = blockIdx.y * 128, bn = blockIdx.x * 128;

  v8f zero = {};
  v8f acc[4][2];
#pragma unroll
  for (int i = 0; i < 4; i++)
#pragma unroll
    for (int j = 0; j < 2; j++) acc[i][j] = zero;

  int arow = tid >> 1, acg = (tid & 1) * 16;   // 128 rows x 32 cols of A tile
  int brow = tid >> 3, bcg = (tid & 7) * 16;   // 32 rows x 128 cols of B tile
  const h16* ag = A + (size_t)(bm + arow) * K + acg;
  const h16* bg = Bm + (size_t)brow * Nc + bn + bcg;

  for (int kt = 0; kt < K; kt += 32) {
    *reinterpret_cast<v16h*>(&As[arow][acg]) = *reinterpret_cast<const v16h*>(ag + kt);
    *reinterpret_cast<v16h*>(&Bs[brow][bcg]) = *reinterpret_cast<const v16h*>(bg + (size_t)kt * Nc);
    if (kt + 32 < K) {                         // warm L2 for the next k-tile
      __builtin_prefetch(ag + kt + 32, 0, 0);
      __builtin_prefetch(bg + (size_t)(kt + 32) * Nc, 0, 0);
    }
    __syncthreads();
    v16h bf[2];
#pragma unroll
    for (int nf = 0; nf < 2; nf++) bf[nf] = load_b16(&Bs[0][wn * 32 + nf * 16], 128, lane);
#pragma unroll
    for (int mf = 0; mf < 4; mf++) {
      v16h af = load_a16(&As[wm * 64 + mf * 16][0], 32, 0, lane);
#pragma unroll
      for (int nf = 0; nf < 2; nf++) acc[mf][nf] = wmma16(af, bf[nf], acc[mf][nf]);
    }
    __syncthreads();
  }

  int mb = (lane >> 4) * 8, nl = lane & 15;
#pragma unroll
  for (int mf = 0; mf < 4; mf++)
#pragma unroll
    for (int nf = 0; nf < 2; nf++)
#pragma unroll
      for (int r = 0; r < 8; r++) {
        int m = bm + wm * 64 + mf * 16 + mb + r;
        int n = bn + wn * 32 + nf * 16 + nl;
        float v = acc[mf][nf][r];
        if (mode == 0) {
          int which = n / Ct, inner = n - which * Ct;
          int hh = inner >> 6, d = inner & 63;
          int bb = m >> 9, q = m & 511;
          h16 hv = (h16)v;
          size_t head = (size_t)bb * Ht + hh;
          if (which == 0)       Qh [(head * Nt + q) * HDt + d] = hv;
          else if (which == 1)  Ktp[(head * HDt + d) * Nt + q] = hv;  // transposed K
          else                  Vh [(head * Nt + q) * HDt + d] = hv;
        } else {
          Cout[(size_t)m * Ct + n] = v + bias[n];
        }
      }
}

// ---------------- attention: one WG per (b, h, 64-query tile) ----------------
// dynamic LDS: S[64][512] f32 (128KB) | Ks[64][512] f16 hd-major (64KB) | Vs[512][64] f16 (64KB)
#define LDS_S_BYTES  (64 * 512 * 4)
#define LDS_K_BYTES  (64 * 512 * 2)
#define LDS_V_BYTES  (512 * 64 * 2)
__global__ __launch_bounds__(256) void attn_kernel(
    const h16* __restrict__ Qh, const h16* __restrict__ Ktp, const h16* __restrict__ Vh,
    const int* __restrict__ mask, float* __restrict__ cls, h16* __restrict__ Oh) {
  extern __shared__ v16h smem_v[];
  char* smem = (char*)smem_v;
  float* S = (float*)smem;                       // 64 x 512
  h16*   Ks = (h16*)(smem + LDS_S_BYTES);        // [hd][key]
  h16*   Vs = Ks + 64 * 512;                     // [key][hd]

  int tid = threadIdx.x, lane = tid & 31, w = tid >> 5;
  int qt = blockIdx.x, hh = blockIdx.y, bb = blockIdx.z;
  size_t head = (size_t)bb * Ht + hh;

  const h16* Kg = Ktp + head * (size_t)(HDt * Nt);
  const h16* Vg = Vh  + head * (size_t)(Nt * HDt);

#if HAVE_TDM
  if (w == 0) {                                  // one DMA issue per tile, wave 0 only
    tdm_load_1d(LDS_S_BYTES,               Kg, LDS_K_BYTES);
    tdm_load_1d(LDS_S_BYTES + LDS_K_BYTES, Vg, LDS_V_BYTES);
  }
#else
#pragma unroll
  for (int i = 0; i < 8; i++) {                  // 2048 v16h chunks each
    int c = tid + i * 256;
    reinterpret_cast<v16h*>(Ks)[c] = reinterpret_cast<const v16h*>(Kg)[c];
    reinterpret_cast<v16h*>(Vs)[c] = reinterpret_cast<const v16h*>(Vg)[c];
  }
#endif

  int r0 = (w >> 1) * 16;                        // wave's 16 query rows (4 groups x 2 col halves)
  const h16* Qg = Qh + (head * Nt + (size_t)qt * 64 + r0) * HDt;
  v16h aq0 = load_a16(Qg, HDt, 0, lane);         // overlaps the TDM transfer
  v16h aq1 = load_a16(Qg, HDt, 32, lane);
#if HAVE_TDM
  if (w == 0) __builtin_amdgcn_s_wait_tensorcnt(0);
#endif
  __syncthreads();

  int mb = (lane >> 4) * 8, nl = lane & 15;
  v8f zero = {};
  const int* mrow = mask + (size_t)bb * Nt * Nt; // key mask = attn_mask[b][0][:]

  // ---- S = scale * Q K^T with key mask ----
  int colbase = (w & 1) * 256;
  for (int nt = 0; nt < 16; nt++) {
    int ncol = colbase + nt * 16;
    v8f acc = zero;
    acc = wmma16(aq0, load_b16(Ks + ncol, 512, lane), acc);
    acc = wmma16(aq1, load_b16(Ks + 32 * 512 + ncol, 512, lane), acc);
    int key = ncol + nl;
    bool valid = mrow[key] != 0;
#pragma unroll
    for (int r = 0; r < 8; r++)
      S[(r0 + mb + r) * 512 + key] = valid ? acc[r] * SCALEF : -1.0e9f;
  }
  __syncthreads();

  // ---- row softmax: 4 lanes per row, 128 cols each (wave32 shuffles) ----
  {
    int row = tid >> 2;
    int c0 = (tid & 3) * 128;
    float* Sr = S + row * 512 + c0;
    float mx = -3.402823466e38f;
    for (int c = 0; c < 128; c++) mx = fmaxf(mx, Sr[c]);
    mx = fmaxf(mx, __shfl_xor(mx, 1, 32));
    mx = fmaxf(mx, __shfl_xor(mx, 2, 32));
    float s = 0.f;
    for (int c = 0; c < 128; c++) { float e = __expf(Sr[c] - mx); Sr[c] = e; s += e; }
    s += __shfl_xor(s, 1, 32);
    s += __shfl_xor(s, 2, 32);
    float inv = 1.f / s;
    bool isrow0 = (qt == 0) && (row == 0);
    for (int c = 0; c < 128; c++) {
      float p = Sr[c] * inv;
      Sr[c] = p;
      if (isrow0) {
        int key = c0 + c;
        if (key >= 1) cls[head * (Nt - 1) + key - 1] = p;   // per-(b,h) row-0 probs
      }
    }
  }
  __syncthreads();

  // ---- O = P V ----
  int hd0 = (w & 1) * 32;
  v8f oacc[2] = {zero, zero};
  for (int kt2 = 0; kt2 < 16; kt2++) {
    v16h ap = load_a32(S + r0 * 512, 512, kt2 * 32, lane);
#pragma unroll
    for (int nf = 0; nf < 2; nf++)
      oacc[nf] = wmma16(ap, load_b16(Vs + kt2 * 32 * 64 + hd0 + nf * 16, 64, lane), oacc[nf]);
  }
#pragma unroll
  for (int nf = 0; nf < 2; nf++)
#pragma unroll
    for (int r = 0; r < 8; r++) {
      int q = qt * 64 + r0 + mb + r;
      int d = hd0 + nf * 16 + nl;
      Oh[((size_t)bb * Nt + q) * Ct + hh * HDt + d] = (h16)oacc[nf][r];
    }
}

// ---------------- top-k (rank-based, stable ties like lax.top_k) ----------------
__global__ __launch_bounds__(256) void topk_kernel(
    const float* __restrict__ cls, const int* __restrict__ mask, float* __restrict__ idx_out) {
  __shared__ float sc[Nt - 1];
  __shared__ int topb[BOUND];
  __shared__ int cnt;
  int b = blockIdx.x, tid = threadIdx.x;
  if (tid == 0) cnt = 0;
  __syncthreads();
  int local = 0;
  for (int j = tid; j < Nt - 1; j += 256) {
    int valid = mask[(size_t)b * Nt * Nt + j + 1] != 0;
    float s = 0.f;
    for (int hh = 0; hh < Ht; hh++) s += cls[((size_t)b * Ht + hh) * (Nt - 1) + j];
    sc[j] = valid ? s * (1.0f / Ht) : -__builtin_inff();
    local += valid;
  }
  atomicAdd(&cnt, local);
  __syncthreads();
  int left = (int)ceilf((float)cnt * 0.7f);
  for (int j = tid; j < Nt - 1; j += 256) {
    float sj = sc[j];
    int rank = 0;
    for (int i = 0; i < Nt - 1; i++) {
      float si = sc[i];
      rank += (si > sj) || (si == sj && i < j);
    }
    if (rank < BOUND) topb[rank] = j;
  }
  __syncthreads();
  for (int p = tid; p < BOUND; p += 256)
    idx_out[(size_t)b * BOUND + p] = (p < left) ? (float)topb[p] : 1.0e9f;
}

// ---------------- broadcast idx over C + BOUNDARY scalar ----------------
__global__ void bcast_kernel(const float* __restrict__ idx, float* __restrict__ indexo,
                             float* __restrict__ bptr) {
  size_t e = (size_t)blockIdx.x * blockDim.x + threadIdx.x;
  size_t total = (size_t)Bt * BOUND * Ct;
  if (e < total) indexo[e] = idx[e / Ct];
  if (e == 0) bptr[0] = (float)BOUND;
}

extern "C" void kernel_launch(void* const* d_in, const int* in_sizes, int n_in,
                              void* d_out, int out_size, void* d_ws, size_t ws_size,
                              hipStream_t stream) {
  (void)in_sizes; (void)n_in; (void)out_size; (void)ws_size;
  const float* x     = (const float*)d_in[0];
  const int*   amask = (const int*)d_in[1];
  const float* wqkv  = (const float*)d_in[2];
  const float* wproj = (const float*)d_in[3];
  const float* bproj = (const float*)d_in[4];

  char* ws = (char*)d_ws;
  size_t off = 0;
  auto alloc = [&](size_t bytes) -> void* {
    void* p = ws + off;
    off += (bytes + 255) & ~(size_t)255;
    return p;
  };
  h16* xh     = (h16*)alloc((size_t)Bt * Nt * Ct * 2);
  h16* wqkvh  = (h16*)alloc((size_t)Ct * 3 * Ct * 2);
  h16* wprojh = (h16*)alloc((size_t)Ct * Ct * 2);
  h16* Qh     = (h16*)alloc((size_t)Bt * Ht * Nt * HDt * 2);
  h16* Ktp    = (h16*)alloc((size_t)Bt * Ht * HDt * Nt * 2);
  h16* Vh     = (h16*)alloc((size_t)Bt * Ht * Nt * HDt * 2);
  float* cls  = (float*)alloc((size_t)Bt * Ht * (Nt - 1) * 4);
  h16* Oh     = xh;  // xh is dead after the QKV GEMM; reuse for attention output

  int nx = Bt * Nt * Ct;
  cvt_f16_kernel<<<(nx + 255) / 256, 256, 0, stream>>>(x, xh, nx);
  int nw0 = Ct * 3 * Ct;
  cvt_f16_kernel<<<(nw0 + 255) / 256, 256, 0, stream>>>(wqkv, wqkvh, nw0);
  int nw1 = Ct * Ct;
  cvt_f16_kernel<<<(nw1 + 255) / 256, 256, 0, stream>>>(wproj, wprojh, nw1);

  gemm_kernel<<<dim3(3 * Ct / 128, Bt * Nt / 128), 256, 0, stream>>>(
      xh, wqkvh, Bt * Nt, 3 * Ct, Ct, 0, Qh, Ktp, Vh, nullptr, nullptr);

  size_t attn_lds = (size_t)LDS_S_BYTES + LDS_K_BYTES + LDS_V_BYTES; // 256KB
  attn_kernel<<<dim3(Nt / 64, Ht, Bt), 256, attn_lds, stream>>>(Qh, Ktp, Vh, amask, cls, Oh);

  float* outp = (float*)d_out;
  gemm_kernel<<<dim3(Ct / 128, Bt * Nt / 128), 256, 0, stream>>>(
      Oh, wprojh, Bt * Nt, Ct, Ct, 1, nullptr, nullptr, nullptr, bproj, outp);

  float* indexp = outp + (size_t)Bt * Nt * Ct;
  float* idxp   = indexp + (size_t)Bt * BOUND * Ct;
  float* bptr   = idxp + (size_t)Bt * BOUND;
  topk_kernel<<<Bt, 256, 0, stream>>>(cls, amask, idxp);
  size_t tot = (size_t)Bt * BOUND * Ct;
  bcast_kernel<<<(unsigned)((tot + 255) / 256), 256, 0, stream>>>(idxp, indexp, bptr);
}
